// PtrNet_49031346651215
// MI455X (gfx1250) — matrix operand order repaired and checked by reading, not compiled, not run on hardware
//
#include <hip/hip_runtime.h>
#include <hip/hip_bf16.h>
#include <math.h>

// ---------------------------------------------------------------------------
// PtrNet on MI455X (gfx1250): bf16 WMMA GEMMs, async global->LDS tile staging
// (GLOBAL_LOAD_ASYNC_TO_LDS_B128, ASYNCcnt), fp32 accumulate.
// B=256, S=64, I=256, H=512, 4H=2048.
// ---------------------------------------------------------------------------

#define BDIM 256
#define SDIM 64
#define IDIM 256
#define HDIM 512
#define G4H  2048

typedef __bf16 bf16;
typedef __attribute__((ext_vector_type(16))) __bf16 v16bf;
typedef __attribute__((ext_vector_type(8)))  float  v8f;

// Low 32 bits of a generic pointer into LDS == LDS byte offset (aperture
// truncation rule, ISA 10.2).
__device__ __forceinline__ unsigned lds_off(const void* p) {
    return (unsigned)(uintptr_t)p;
}

__device__ __forceinline__ void async_load_b128(unsigned lds_addr, const void* gptr) {
    // GLOBAL_LOAD_ASYNC_TO_LDS_B128: VDST = per-lane LDS byte offset,
    // VADDR = per-lane 64-bit global address. Tracked by ASYNCcnt.
    asm volatile("global_load_async_to_lds_b128 %0, %1, off"
                 :: "v"(lds_addr), "v"((unsigned long long)(uintptr_t)gptr)
                 : "memory");
}

__device__ __forceinline__ void wait_async0() {
    asm volatile("s_wait_asynccnt 0x0" ::: "memory");
}

// ---------------------------------------------------------------------------
// GEMM: C[M,N] = A0[M,K0]*W0[N,K0]^T + A1[M,K1]*W1[N,K1]^T + bias0[N] + bias1[N]
// A, W row-major with K contiguous. Block tile 64x64, 128 threads (4 waves),
// wave tile 32x32 (2x2 WMMA frags). Double-buffered async LDS staging.
// Requires K0, K1 multiples of 64 (even #tiles -> safe buffer parity at the
// phase boundary). Here K ∈ {256, 512}.
// ---------------------------------------------------------------------------
#define TBM 64
#define TBN 64
#define TBK 32

__global__ __launch_bounds__(128) void gemm_bt_wmma_kernel(
    const float* __restrict__ A0, int lda0, int K0,
    const float* __restrict__ W0, int ldw0,
    const float* __restrict__ A1, int lda1, int K1,
    const float* __restrict__ W1, int ldw1,
    const float* __restrict__ bias0, const float* __restrict__ bias1,
    float* __restrict__ C, int ldc)
{
    __shared__ float Asf[2][TBM][TBK];   // fp32 staged tiles (async copy target)
    __shared__ float Bsf[2][TBN][TBK];

    const int tid  = threadIdx.x;
    const int lane = tid & 31;
    const int wave = tid >> 5;          // 0..3
    const int wm   = wave & 1;
    const int wn   = wave >> 1;
    const int half = lane >> 4;         // 0: lanes0-15, 1: lanes16-31
    const int l16  = lane & 15;

    const int block_m = blockIdx.y * TBM;
    const int block_n = blockIdx.x * TBN;

    v8f acc[2][2] = {};

    // per-thread tile-copy mapping: 512 float4 per tile / 128 threads = 4 each
    for (int phase = 0; phase < 2; ++phase) {
        const float* A  = phase ? A1  : A0;
        const float* W  = phase ? W1  : W0;
        const int    K  = phase ? K1  : K0;
        const int   lda = phase ? lda1 : lda0;
        const int   ldw = phase ? ldw1 : ldw0;
        if (A == nullptr) continue;

        // prologue: stage first K tile into buffer 0
        #pragma unroll
        for (int c = 0; c < 4; ++c) {
            const int f4  = tid + c * 128;
            const int row = f4 >> 3;
            const int col = (f4 & 7) << 2;
            async_load_b128(lds_off(&Asf[0][row][col]),
                            A + (size_t)(block_m + row) * lda + col);
            async_load_b128(lds_off(&Bsf[0][row][col]),
                            W + (size_t)(block_n + row) * ldw + col);
        }

        for (int k0 = 0; k0 < K; k0 += TBK) {
            const int buf = (k0 / TBK) & 1;
            wait_async0();
            __syncthreads();

            if (k0 + TBK < K) {      // stage next K tile into the other buffer
                #pragma unroll
                for (int c = 0; c < 4; ++c) {
                    const int f4  = tid + c * 128;
                    const int row = f4 >> 3;
                    const int col = (f4 & 7) << 2;
                    async_load_b128(lds_off(&Asf[buf ^ 1][row][col]),
                                    A + (size_t)(block_m + row) * lda + k0 + TBK + col);
                    async_load_b128(lds_off(&Bsf[buf ^ 1][row][col]),
                                    W + (size_t)(block_n + row) * ldw + k0 + TBK + col);
                }
            }

            // --- build bf16 fragments (hardware cvt) per ISA 7.12.2 layouts
            v16bf afrag[2], bfrag[2];
            #pragma unroll
            for (int i = 0; i < 2; ++i) {
                // A 16x32: lane half selects K {0..7,16..23} vs {8..15,24..31}
                const float* arow = &Asf[buf][wm * 32 + i * 16 + l16][0];
                #pragma unroll
                for (int e = 0; e < 8; ++e) {
                    afrag[i][e]     = (bf16)arow[half * 8 + e];
                    afrag[i][8 + e] = (bf16)arow[16 + half * 8 + e];
                }
            }
            #pragma unroll
            for (int j = 0; j < 2; ++j) {
                // B 32x16: lane half selects K {0..7,16..23} vs {8..15,24..31}, N = lane%16
                const float* brow = &Bsf[buf][wn * 32 + j * 16 + l16][0];
                #pragma unroll
                for (int e = 0; e < 8; ++e) {
                    bfrag[j][e]     = (bf16)brow[half * 8 + e];
                    bfrag[j][8 + e] = (bf16)brow[16 + half * 8 + e];
                }
            }

            #pragma unroll
            for (int i = 0; i < 2; ++i)
                #pragma unroll
                for (int j = 0; j < 2; ++j)
                    acc[i][j] = __builtin_amdgcn_wmma_f32_16x16x32_bf16(
                        false, afrag[i], false, bfrag[j], (short)0, acc[i][j], false, false);
        }
    }

    // --- epilogue: f32 C layout (VGPR r: M=r lanes0-15, M=r+8 lanes16-31; N=lane%16)
    #pragma unroll
    for (int j = 0; j < 2; ++j) {
        const int gn = block_n + wn * 32 + j * 16 + l16;
        float bv = 0.f;
        if (bias0) bv += bias0[gn];
        if (bias1) bv += bias1[gn];
        #pragma unroll
        for (int i = 0; i < 2; ++i) {
            const int gm_base = block_m + wm * 32 + i * 16 + (half << 3);
            #pragma unroll
            for (int r = 0; r < 8; ++r)
                C[(size_t)(gm_base + r) * ldc + gn] = acc[i][j][r] + bv;
        }
    }
}

// ---------------------------------------------------------------------------
// LSTM gate fusion: gates[B,4H] (i,f,g,o) + c_in -> h_out, c_out (+enc_c store)
// ---------------------------------------------------------------------------
__global__ __launch_bounds__(256) void lstm_eltwise_kernel(
    const float* __restrict__ gates, const float* __restrict__ c_in,
    float* __restrict__ h_out, float* __restrict__ c_out,
    float* __restrict__ c_store)
{
    const int idx = blockIdx.x * 256 + threadIdx.x;     // 0..B*H-1
    const int b = idx >> 9;
    const int h = idx & (HDIM - 1);
    const size_t g0 = (size_t)b * G4H + h;
    const float gi = gates[g0];
    const float gf = gates[g0 + HDIM];
    const float gg = gates[g0 + 2 * HDIM];
    const float go = gates[g0 + 3 * HDIM];
    const float i  = 1.f / (1.f + __expf(-gi));
    const float f  = 1.f / (1.f + __expf(-gf));
    const float o  = 1.f / (1.f + __expf(-go));
    const float c2 = f * c_in[idx] + i * tanhf(gg);
    const float h2 = o * tanhf(c2);
    c_out[idx] = c2;
    h_out[idx] = h2;
    if (c_store) c_store[idx] = c2;
}

// ---------------------------------------------------------------------------
// Attention: per batch element b (one workgroup of 256):
//   u[s] = sum_h V[h]*tanh(ci[b,h]+e[s,b,h]) + Vb ; a = softmax(u) ;
//   d2[b,h] = sum_s a[s]*enc_c[s,b,h] ; dists[b,s] = a[s]
// ---------------------------------------------------------------------------
__global__ __launch_bounds__(256) void attention_kernel(
    const float* __restrict__ ci,      // [B,H]
    const float* __restrict__ e,       // [S,B,H]
    const float* __restrict__ enc_c,   // [S,B,H]
    const float* __restrict__ Vw, const float* __restrict__ Vb,
    float* __restrict__ dists_step,    // [B,S]
    float* __restrict__ d_next)        // [B,H]
{
    __shared__ float ci_sh[HDIM];
    __shared__ float v_sh[HDIM];
    __shared__ float part_sh[256];
    __shared__ float u_sh[SDIM];
    __shared__ float a_sh[SDIM];

    const int b   = blockIdx.x;
    const int tid = threadIdx.x;

    ci_sh[tid]       = ci[(size_t)b * HDIM + tid];
    ci_sh[tid + 256] = ci[(size_t)b * HDIM + 256 + tid];
    v_sh[tid]        = Vw[tid];
    v_sh[tid + 256]  = Vw[256 + tid];
    __syncthreads();

    // 4 threads cooperate per source position s
    const int sp  = tid >> 2;
    const int sub = tid & 3;
    const float* erow = e + ((size_t)sp * BDIM + b) * HDIM;
    float p = 0.f;
    for (int h = sub; h < HDIM; h += 4)
        p += v_sh[h] * tanhf(ci_sh[h] + erow[h]);
    part_sh[tid] = p;
    __syncthreads();
    if (sub == 0)
        u_sh[sp] = part_sh[tid] + part_sh[tid + 1] + part_sh[tid + 2] + part_sh[tid + 3] + Vb[0];
    __syncthreads();

    // softmax over 64 (redundant per-thread scan is cheap)
    float mx = -3.4e38f;
    #pragma unroll 8
    for (int s = 0; s < SDIM; ++s) mx = fmaxf(mx, u_sh[s]);
    float den = 0.f;
    #pragma unroll 8
    for (int s = 0; s < SDIM; ++s) den += __expf(u_sh[s] - mx);
    if (tid < SDIM) {
        const float a = __expf(u_sh[tid] - mx) / den;
        a_sh[tid] = a;
        dists_step[(size_t)b * SDIM + tid] = a;
    }
    __syncthreads();

    // d2 = sum_s a[s] * enc_c[s,b,:]
    #pragma unroll
    for (int hh = 0; hh < 2; ++hh) {
        const int h = tid + hh * 256;
        float acc = 0.f;
        for (int s = 0; s < SDIM; ++s)
            acc += a_sh[s] * enc_c[((size_t)s * BDIM + b) * HDIM + h];
        d_next[(size_t)b * HDIM + h] = acc;
    }
}

__global__ void fill_kernel(float* __restrict__ p, float v, int n) {
    const int i = blockIdx.x * 256 + threadIdx.x;
    if (i < n) p[i] = v;
}

// ---------------------------------------------------------------------------
extern "C" void kernel_launch(void* const* d_in, const int* in_sizes, int n_in,
                              void* d_out, int out_size, void* d_ws, size_t ws_size,
                              hipStream_t stream) {
    const float* input = (const float*)d_in[0];   // [S,B,I]
    const float* h0    = (const float*)d_in[1];   // [B,H]
    const float* c0    = (const float*)d_in[2];   // [B,H]
    const float* eWih  = (const float*)d_in[3];   // [S,4H,I]
    const float* eWhh  = (const float*)d_in[4];   // [S,4H,H]
    const float* ebih  = (const float*)d_in[5];   // [S,4H]
    const float* ebhh  = (const float*)d_in[6];   // [S,4H]
    const float* dWih  = (const float*)d_in[7];   // [S,4H,H]
    const float* dWhh  = (const float*)d_in[8];   // [S,4H,H]
    const float* dbih  = (const float*)d_in[9];
    const float* dbhh  = (const float*)d_in[10];
    const float* Wew   = (const float*)d_in[11];  // [H,H]
    const float* Web   = (const float*)d_in[12];  // [H]
    const float* Wdw   = (const float*)d_in[13];  // [H,H]
    const float* Wdb   = (const float*)d_in[14];  // [H]
    const float* Vw    = (const float*)d_in[15];  // [1,H]
    const float* Vb    = (const float*)d_in[16];  // [1]
    float* out = (float*)d_out;                   // [S,B,S]

    const size_t BH = (size_t)BDIM * HDIM;        // 131072
    float* ws    = (float*)d_ws;
    float* hbuf  = ws;                 // 2*BH
    float* cbuf  = hbuf  + 2 * BH;     // 2*BH
    float* dbuf  = cbuf  + 2 * BH;     // 2*BH
    float* gates = dbuf  + 2 * BH;     // 4*BH  ([B,4H])
    float* cibuf = gates + 4 * BH;     // BH
    float* enc_c = cibuf + BH;         // 64*BH
    float* ebuf  = enc_c + SDIM * BH;  // 64*BH

    const dim3 gblk(128);

    // ---------------- encoder ----------------
    const float* hin = h0;
    const float* cin = c0;
    for (int s = 0; s < SDIM; ++s) {
        float* hout = hbuf + (size_t)(s & 1) * BH;
        float* cout = cbuf + (size_t)(s & 1) * BH;
        gemm_bt_wmma_kernel<<<dim3(G4H / TBN, BDIM / TBM), gblk, 0, stream>>>(
            input + (size_t)s * BDIM * IDIM, IDIM, IDIM,
            eWih  + (size_t)s * G4H * IDIM, IDIM,
            hin, HDIM, HDIM,
            eWhh  + (size_t)s * G4H * HDIM, HDIM,
            ebih + (size_t)s * G4H, ebhh + (size_t)s * G4H,
            gates, G4H);
        lstm_eltwise_kernel<<<(int)(BH / 256), 256, 0, stream>>>(
            gates, cin, hout, cout, enc_c + (size_t)s * BH);
        hin = hout; cin = cout;
    }

    // ---------------- e = enc_c @ We^T + We_b  (M=S*B, N=H, K=H) ----------------
    gemm_bt_wmma_kernel<<<dim3(HDIM / TBN, (SDIM * BDIM) / TBM), gblk, 0, stream>>>(
        enc_c, HDIM, HDIM, Wew, HDIM,
        nullptr, 0, 0, nullptr, 0,
        Web, nullptr, ebuf, HDIM);

    // ---------------- decoder ----------------
    fill_kernel<<<(int)(BH / 256), 256, 0, stream>>>(dbuf, -1.0f, (int)BH);
    const float* din = dbuf;
    for (int s = 0; s < SDIM; ++s) {
        float* hout = hbuf + (size_t)(s & 1) * BH;
        float* cout = cbuf + (size_t)(s & 1) * BH;
        float* dout = dbuf + (size_t)((s & 1) ^ 1) * BH;
        gemm_bt_wmma_kernel<<<dim3(G4H / TBN, BDIM / TBM), gblk, 0, stream>>>(
            din, HDIM, HDIM,
            dWih + (size_t)s * G4H * HDIM, HDIM,
            hin, HDIM, HDIM,
            dWhh + (size_t)s * G4H * HDIM, HDIM,
            dbih + (size_t)s * G4H, dbhh + (size_t)s * G4H,
            gates, G4H);
        lstm_eltwise_kernel<<<(int)(BH / 256), 256, 0, stream>>>(
            gates, cin, hout, cout, nullptr);
        // ci = c2 @ Wd^T + Wd_b
        gemm_bt_wmma_kernel<<<dim3(HDIM / TBN, BDIM / TBM), gblk, 0, stream>>>(
            cout, HDIM, HDIM, Wdw, HDIM,
            nullptr, 0, 0, nullptr, 0,
            Wdb, nullptr, cibuf, HDIM);
        attention_kernel<<<BDIM, 256, 0, stream>>>(
            cibuf, ebuf, enc_c, Vw, Vb,
            out + (size_t)s * BDIM * SDIM, dout);
        hin = hout; cin = cout; din = dout;
    }
}
